// SanaAttention_46325517254736
// MI455X (gfx1250) — matrix operand order repaired and checked by reading, not compiled, not run on hardware
//
#include <hip/hip_runtime.h>
#include <hip/hip_bf16.h>

// ---------------------------------------------------------------------------
// SanaAttention forward for MI455X (gfx1250, wave32, WMMA).
// B=4, C=1024, H=W=64 (N=4096), HEADS=128 (per branch), HEAD_DIM=8.
// GEMMs run on v_wmma_f32_16x16x32_bf16; everything else VALU.
// ---------------------------------------------------------------------------

typedef __attribute__((ext_vector_type(16))) __bf16 v16bf;
typedef __attribute__((ext_vector_type(8)))  __bf16 v8bf;
typedef __attribute__((ext_vector_type(8)))  float  v8f;

#define BB   4
#define CC   1024
#define NN   4096            // H*W
#define C3   3072            // 3*C
#define HEADS2 256           // 2*HEADS
#define EPS_LIN 1e-15f
#define EPS_BN  1e-5f

// ---------------------------------------------------------------------------
// f32 -> bf16 elementwise convert (weights)
// ---------------------------------------------------------------------------
__global__ void k_f32_to_bf16(const float* __restrict__ s, __bf16* __restrict__ d,
                              long long n) {
  long long i = (long long)blockIdx.x * blockDim.x + threadIdx.x;
  if (i < n) d[i] = (__bf16)s[i];
}

// ---------------------------------------------------------------------------
// Transpose + convert: src f32 [batch][rows][cols] -> dst bf16 [batch][cols][rows]
// 32x32 LDS tiles, coalesced both sides.
// ---------------------------------------------------------------------------
__global__ void k_transpose_f32_bf16(const float* __restrict__ src,
                                     __bf16* __restrict__ dst,
                                     int rows, int cols,
                                     long long sstride, long long dstride) {
  __shared__ float tile[32][33];
  const float* s = src + (long long)blockIdx.z * sstride;
  __bf16*      d = dst + (long long)blockIdx.z * dstride;
  int c0 = blockIdx.x * 32, r0 = blockIdx.y * 32;
  int tx = threadIdx.x, ty = threadIdx.y;
  #pragma unroll
  for (int i = ty; i < 32; i += 8)
    tile[i][tx] = s[(long long)(r0 + i) * cols + c0 + tx];
  __syncthreads();
  #pragma unroll
  for (int i = ty; i < 32; i += 8)
    d[(long long)(c0 + i) * rows + r0 + tx] = (__bf16)tile[tx][i];
}

// ---------------------------------------------------------------------------
// WMMA GEMM:  C[M x N] (f32, row-major) = A[M x K] (bf16 row-major)
//                                         x B[K x N], B given transposed:
//             BT [N x K] bf16 row-major  (so B-fragments load contiguously)
// One wave per block; wave computes a 32x64 tile (2x4 WMMA tiles).
// grid = (N/64, M/32, batch)
// ---------------------------------------------------------------------------
__global__ __launch_bounds__(32)
void k_wmma_gemm_bf16(const __bf16* __restrict__ A,
                      const __bf16* __restrict__ BT,
                      float* __restrict__ C,
                      int N, int K,
                      long long strideBT, long long strideC) {
  int lane = threadIdx.x;
  int half = lane >> 4;      // 0 | 1
  int l16  = lane & 15;
  const __bf16* Bp = BT + (long long)blockIdx.z * strideBT;
  float*        Cp = C  + (long long)blockIdx.z * strideC;
  int n0 = blockIdx.x * 64;
  int m0 = blockIdx.y * 32;

  v8f acc[2][4] = {};

  for (int k0 = 0; k0 < K; k0 += 32) {
    v16bf a[2], b[4];
    // A fragment: lane holds row (m0+mi*16+l16), K = k0 + half*16 + {0..15}
    #pragma unroll
    for (int mi = 0; mi < 2; ++mi) {
      const __bf16* ap = A + (long long)(m0 + mi * 16 + l16) * K + k0 + half * 16;
      v8bf lo = *(const v8bf*)ap;
      v8bf hi = *(const v8bf*)(ap + 8);
      a[mi] = __builtin_shufflevector(lo, hi, 0,1,2,3,4,5,6,7,8,9,10,11,12,13,14,15);
      // next K slab, WGP-scope prefetch (locality 3 -> near caches; consumed
      // by this same wave 1 iteration later)
      __builtin_prefetch(ap + 32, 0, 3);
    }
    // B fragment: lane holds col (n0+ni*16+l16), K = k0 + half*16 + {0..15}
    #pragma unroll
    for (int ni = 0; ni < 4; ++ni) {
      const __bf16* bp = Bp + (long long)(n0 + ni * 16 + l16) * K + k0 + half * 16;
      b[ni] = *(const v16bf*)bp;
      __builtin_prefetch(bp + 32, 0, 3);
    }
    #pragma unroll
    for (int mi = 0; mi < 2; ++mi)
      #pragma unroll
      for (int ni = 0; ni < 4; ++ni)
        acc[mi][ni] = __builtin_amdgcn_wmma_f32_16x16x32_bf16(
            false, a[mi], false, b[ni], (short)0, acc[mi][ni], false, false);
  }

  // C/D layout: VGPR r <-> row (half*8 + r), col = l16
  #pragma unroll
  for (int mi = 0; mi < 2; ++mi)
    #pragma unroll
    for (int ni = 0; ni < 4; ++ni)
      #pragma unroll
      for (int r = 0; r < 8; ++r) {
        int row = m0 + mi * 16 + half * 8 + r;
        int col = n0 + ni * 16 + l16;
        Cp[(long long)row * N + col] = acc[mi][ni][r];
      }
}

// ---------------------------------------------------------------------------
// Depthwise 5x5 SAME conv, groups = 3C.  in/out: [B][3C][64][64]
// ---------------------------------------------------------------------------
__global__ void k_dwconv5(const float* __restrict__ in, const float* __restrict__ w,
                          float* __restrict__ out) {
  long long idx = (long long)blockIdx.x * 256 + threadIdx.x;   // B*3C*N
  int n  = (int)(idx & 4095);
  long long bc = idx >> 12;          // b*3C + ch
  int ch = (int)(bc % C3);
  int y = n >> 6, x = n & 63;
  const float* ip = in + (bc << 12);
  const float* wp = w + ch * 25;
  float acc = 0.f;
  #pragma unroll
  for (int dy = 0; dy < 5; ++dy) {
    int iy = y + dy - 2;
    if (iy < 0 || iy > 63) continue;
    #pragma unroll
    for (int dx = 0; dx < 5; ++dx) {
      int ix = x + dx - 2;
      if (ix < 0 || ix > 63) continue;
      acc += ip[iy * 64 + ix] * wp[dy * 5 + dx];
    }
  }
  out[idx] = acc;
}

// ---------------------------------------------------------------------------
// Grouped 1x1 conv: groups = 384, 8-in -> 8-out per group.
// ---------------------------------------------------------------------------
__global__ void k_pwconv(const float* __restrict__ in, const float* __restrict__ w,
                         float* __restrict__ out) {
  long long idx = (long long)blockIdx.x * 256 + threadIdx.x;   // B*3C*N
  int n  = (int)(idx & 4095);
  long long bc = idx >> 12;
  int oc = (int)(bc % C3);
  long long b = bc / C3;
  int g = oc >> 3;
  const float* ip = in + (((b * C3 + (long long)g * 8) << 12) + n);
  const float* wp = w + oc * 8;
  float acc = 0.f;
  #pragma unroll
  for (int i = 0; i < 8; ++i) acc += ip[(long long)i << 12] * wp[i];
  out[idx] = acc;
}

// ---------------------------------------------------------------------------
// Linear attention per (b, head): head channels = hs[ht*24 .. ht*24+24)
//   q' = rows 0..7 (relu), k' = rows 8..15 (relu), v' = rows 16..23,
//   v padded with a row of ones.  scores[d][e] = sum_n v[d][n] k[e][n]  (9x8)
//   out[d][n] = sum_e scores[d][e] q[e][n];  att = out[0..7] / (out[8]+eps)
// grid = (256 heads, 4 batch), block = 256 (8 waves, shuffle reduce).
// ---------------------------------------------------------------------------
__global__ __launch_bounds__(256)
void k_linear_attention(const float* __restrict__ qkv, const float* __restrict__ ms,
                        float* __restrict__ att) {
  int ht = blockIdx.x, b = blockIdx.y;
  int tid = threadIdx.x, lane = tid & 31, wv = tid >> 5;
  const float* src = (ht < 128)
      ? qkv + (((long long)b * C3 + (long long)ht * 24) << 12)
      : ms  + (((long long)b * C3 + (long long)(ht - 128) * 24) << 12);

  float acc[9][8];
  #pragma unroll
  for (int d = 0; d < 9; ++d)
    #pragma unroll
    for (int e = 0; e < 8; ++e) acc[d][e] = 0.f;

  for (int n = tid; n < NN; n += 256) {
    float kv[8], vv[8];
    #pragma unroll
    for (int e = 0; e < 8; ++e) kv[e] = fmaxf(src[(long long)(8 + e) * NN + n], 0.f);
    #pragma unroll
    for (int d = 0; d < 8; ++d) vv[d] = src[(long long)(16 + d) * NN + n];
    #pragma unroll
    for (int e = 0; e < 8; ++e) {
      float ke = kv[e];
      #pragma unroll
      for (int d = 0; d < 8; ++d) acc[d][e] += vv[d] * ke;
      acc[8][e] += ke;                      // ones-padded v row
    }
  }

  __shared__ float partial[8][72];
  __shared__ float sc[72];
  #pragma unroll
  for (int i = 0; i < 72; ++i) {
    float v = acc[i / 8][i % 8];
    #pragma unroll
    for (int off = 16; off; off >>= 1) v += __shfl_xor(v, off, 32);
    if (lane == 0) partial[wv][i] = v;
  }
  __syncthreads();
  if (tid < 72) {
    float v = 0.f;
    #pragma unroll
    for (int w = 0; w < 8; ++w) v += partial[w][tid];
    sc[tid] = v;
  }
  __syncthreads();

  for (int n = tid; n < NN; n += 256) {
    float qv[8];
    #pragma unroll
    for (int e = 0; e < 8; ++e) qv[e] = fmaxf(src[(long long)e * NN + n], 0.f);
    float den = EPS_LIN;
    #pragma unroll
    for (int e = 0; e < 8; ++e) den += sc[64 + e] * qv[e];
    float inv = 1.f / den;
    #pragma unroll
    for (int d = 0; d < 8; ++d) {
      float o = 0.f;
      #pragma unroll
      for (int e = 0; e < 8; ++e) o += sc[d * 8 + e] * qv[e];
      att[(((long long)b * 2048 + (long long)ht * 8 + d) << 12) + n] = o * inv;
    }
  }
}

// ---------------------------------------------------------------------------
// BatchNorm stats: per channel c, sum & sumsq over B*N = 16384 values.
// grid = 1024, block = 256.
// ---------------------------------------------------------------------------
__global__ __launch_bounds__(256)
void k_bn_stats(const float* __restrict__ y, float* __restrict__ stats) {
  int c = blockIdx.x, tid = threadIdx.x, lane = tid & 31, wv = tid >> 5;
  float s = 0.f, ss = 0.f;
  for (int i = tid; i < BB * NN; i += 256) {
    int b = i >> 12, n = i & 4095;
    float v = y[(((long long)b * CC + c) << 12) + n];
    s += v; ss += v * v;
  }
  #pragma unroll
  for (int off = 16; off; off >>= 1) {
    s  += __shfl_xor(s,  off, 32);
    ss += __shfl_xor(ss, off, 32);
  }
  __shared__ float ps[8], pss[8];
  if (lane == 0) { ps[wv] = s; pss[wv] = ss; }
  __syncthreads();
  if (tid == 0) {
    float S = 0.f, SS = 0.f;
    #pragma unroll
    for (int w = 0; w < 8; ++w) { S += ps[w]; SS += pss[w]; }
    stats[c] = S; stats[CC + c] = SS;
  }
}

// ---------------------------------------------------------------------------
// BatchNorm apply (population var, affine).
// ---------------------------------------------------------------------------
__global__ void k_bn_apply(const float* __restrict__ y, const float* __restrict__ stats,
                           const float* __restrict__ gamma, const float* __restrict__ beta,
                           float* __restrict__ out) {
  long long idx = (long long)blockIdx.x * 256 + threadIdx.x;   // B*C*N
  int c = (int)((idx >> 12) & (CC - 1));
  const float invCnt = 1.f / (float)(BB * NN);
  float mean = stats[c] * invCnt;
  float var  = stats[CC + c] * invCnt - mean * mean;
  out[idx] = (y[idx] - mean) * rsqrtf(var + EPS_BN) * gamma[c] + beta[c];
}

// ---------------------------------------------------------------------------
// Host launch
// ---------------------------------------------------------------------------
extern "C" void kernel_launch(void* const* d_in, const int* in_sizes, int n_in,
                              void* d_out, int out_size, void* d_ws, size_t ws_size,
                              hipStream_t stream) {
  (void)in_sizes; (void)n_in; (void)out_size; (void)ws_size;
  const float* x     = (const float*)d_in[0];
  const float* Wq    = (const float*)d_in[1];
  const float* Wk    = (const float*)d_in[2];
  const float* Wv    = (const float*)d_in[3];
  const float* dw_w  = (const float*)d_in[4];
  const float* pw_w  = (const float*)d_in[5];
  const float* Wout  = (const float*)d_in[6];
  const float* gamma = (const float*)d_in[7];
  const float* beta  = (const float*)d_in[8];
  float* out = (float*)d_out;

  // workspace carve-up (~875 MB)
  char* ws = (char*)d_ws;
  auto alloc = [&](size_t bytes) -> void* {
    void* p = (void*)ws;
    ws += (bytes + 255) & ~(size_t)255;
    return p;
  };
  __bf16* xT      = (__bf16*)alloc((size_t)BB * NN * CC * 2);          // [B][N][C]
  __bf16* Wq_bf   = (__bf16*)alloc((size_t)CC * CC * 2);
  __bf16* Wk_bf   = (__bf16*)alloc((size_t)CC * CC * 2);
  __bf16* Wv_bf   = (__bf16*)alloc((size_t)CC * CC * 2);
  __bf16* Wout_bf = (__bf16*)alloc((size_t)CC * 2 * CC * 2);
  float*  qkv     = (float*) alloc((size_t)BB * C3 * NN * 4);          // [B][3C][N]
  float*  ms1     = (float*) alloc((size_t)BB * C3 * NN * 4);
  float*  msb     = (float*) alloc((size_t)BB * C3 * NN * 4);
  float*  att     = (float*) alloc((size_t)BB * 2 * CC * NN * 4);      // [B][2C][N]
  __bf16* attT    = (__bf16*)alloc((size_t)BB * NN * 2 * CC * 2);      // [B][N][2C]
  float*  y       = (float*) alloc((size_t)BB * CC * NN * 4);          // [B][C][N]
  float*  stats   = (float*) alloc((size_t)2 * CC * 4);

  // 1) weights -> bf16
  k_f32_to_bf16<<<(CC * CC) / 256, 256, 0, stream>>>(Wq, Wq_bf, (long long)CC * CC);
  k_f32_to_bf16<<<(CC * CC) / 256, 256, 0, stream>>>(Wk, Wk_bf, (long long)CC * CC);
  k_f32_to_bf16<<<(CC * CC) / 256, 256, 0, stream>>>(Wv, Wv_bf, (long long)CC * CC);
  k_f32_to_bf16<<<(CC * 2 * CC) / 256, 256, 0, stream>>>(Wout, Wout_bf, (long long)CC * 2 * CC);

  // 2) x [B][C][N] -> xT bf16 [B][N][C]
  k_transpose_f32_bf16<<<dim3(NN / 32, CC / 32, BB), dim3(32, 8), 0, stream>>>(
      x, xT, CC, NN, (long long)CC * NN, (long long)NN * CC);

  // 3) QKV projections (WMMA): qkv[b][sel*C + co][n] = W[co,:] . x[b,:,n]
  const __bf16* Wsel[3] = {Wq_bf, Wk_bf, Wv_bf};
  for (int sel = 0; sel < 3; ++sel)
    k_wmma_gemm_bf16<<<dim3(NN / 64, CC / 32, BB), 32, 0, stream>>>(
        Wsel[sel], xT, qkv + (long long)sel * CC * NN,
        NN, CC, (long long)NN * CC, (long long)C3 * NN);

  // 4) depthwise 5x5, 5) grouped 1x1
  long long convElems = (long long)BB * C3 * NN;
  k_dwconv5<<<(unsigned)(convElems / 256), 256, 0, stream>>>(qkv, dw_w, ms1);
  k_pwconv <<<(unsigned)(convElems / 256), 256, 0, stream>>>(ms1, pw_w, msb);

  // 6) linear attention -> att [B][2C][N]
  k_linear_attention<<<dim3(HEADS2, BB), 256, 0, stream>>>(qkv, msb, att);

  // 7) att -> attT bf16 [B][N][2C]
  k_transpose_f32_bf16<<<dim3(NN / 32, (2 * CC) / 32, BB), dim3(32, 8), 0, stream>>>(
      att, attT, 2 * CC, NN, (long long)2 * CC * NN, (long long)NN * 2 * CC);

  // 8) output projection (WMMA): y[b][co][n] = Wout[co,:] . att[b,:,n]
  k_wmma_gemm_bf16<<<dim3(NN / 64, CC / 32, BB), 32, 0, stream>>>(
      Wout_bf, attT, y, NN, 2 * CC, (long long)NN * 2 * CC, (long long)CC * NN);

  // 9) batch-norm
  k_bn_stats<<<CC, 256, 0, stream>>>(y, stats);
  long long outElems = (long long)BB * CC * NN;
  k_bn_apply<<<(unsigned)(outElems / 256), 256, 0, stream>>>(y, stats, gamma, beta, out);
}
